// MODEL_57690000720177
// MI455X (gfx1250) — matrix operand order repaired and checked by reading, not compiled, not run on hardware
//
#include <hip/hip_runtime.h>
#include <hip/hip_bf16.h>

typedef __attribute__((ext_vector_type(16))) _Float16 v16h;
typedef __attribute__((ext_vector_type(4)))  _Float16 v4h;
typedef __attribute__((ext_vector_type(8)))  float    v8f;
typedef _Float16 f16;

#define S_LEN 1024
#define B_SZ  256
#define K_DIM 128
#define QA_DIM 256
#define M_SZ  64
#define V_DIM 256
#define F_DIM 128
#define CAT_DIM (V_DIM + F_DIM)       // 384
#define SB    (S_LEN * B_SZ)          // 262144 rows
#define NROWT (SB / 16)               // 16384 row tiles

__device__ __forceinline__ v8f wmma16(v16h a, v16h b, v8f c) {
    // D = A(16x32 f16) * B(32x16 f16) + C(16x16 f32)
    return __builtin_amdgcn_wmma_f32_16x16x32_f16(
        false, a, false, b, (short)0, c, false, false);
}

// A-fragment (16x32 f16, row-major source). ISA 7.12.2:
// lane: row = lane%16; kb = (lane<16?0:8); a[i]=A[row][k0+kb+i], a[8+i]=A[row][k0+16+kb+i]
__device__ __forceinline__ v16h load_afrag(const f16* A, int lda, int row0, int k0, int lane) {
    const f16* p = A + (size_t)(row0 + (lane & 15)) * lda + k0;
    int kb = (lane >> 4) * 8;
    v16h a;
#pragma unroll
    for (int i = 0; i < 8; ++i) { a[i] = p[kb + i]; a[8 + i] = p[16 + kb + i]; }
    return a;
}

// B-fragment (32x16) from PRE-TRANSPOSED f16 weights Bt[N][Kld]:
// lane: col = lane%16; kb = (lane<16?0:8); b[j]=Bt[col][k0+kb+j], b[8+j]=Bt[col][k0+16+kb+j].
__device__ __forceinline__ v16h load_bfragT(const f16* Bt, int ldk, int k0, int col0, int lane) {
    const f16* p = Bt + (size_t)(col0 + (lane & 15)) * ldk + k0 + (lane >> 4) * 8;
    v16h b;
#pragma unroll
    for (int j = 0; j < 8; ++j) { b[j] = p[j]; b[8 + j] = p[16 + j]; }
    return b;
}

// ---------- Kernel 0: convert + transpose weights to f16 (one-shot, tiny) ----------
__global__ void kPrepW(const float* __restrict__ key, const float* __restrict__ eW,
                       const float* __restrict__ aW,  const float* __restrict__ rW,
                       f16* __restrict__ keyh, f16* __restrict__ eWt,
                       f16* __restrict__ aWt,  f16* __restrict__ rWt) {
    int idx = blockIdx.x * 256 + threadIdx.x;
    if (idx < M_SZ * K_DIM) keyh[idx] = (f16)key[idx];              // already [N][K]
    if (idx < V_DIM * QA_DIM) {                                     // [K][N] -> [N][K]
        int n = idx >> 8, k = idx & 255;
        eWt[n * QA_DIM + k] = (f16)eW[(size_t)k * V_DIM + n];
        aWt[n * QA_DIM + k] = (f16)aW[(size_t)k * V_DIM + n];
    }
    if (idx < F_DIM * CAT_DIM) {                                    // [K][N] -> [N][K]
        int n = idx / CAT_DIM, k = idx % CAT_DIM;
        rWt[(size_t)n * CAT_DIM + k] = (f16)rW[(size_t)k * F_DIM + n];
    }
}

// ---------------- Kernel 1: embedding gather + f32->f16 (vectorized) ----------------
__global__ void kEmbed(const int* __restrict__ qd, const int* __restrict__ qad,
                       const float* __restrict__ qT, const float* __restrict__ qaT,
                       f16* __restrict__ qE, f16* __restrict__ qaE) {
    size_t row = blockIdx.x;             // row = s*B + b
    int t = threadIdx.x;                 // 64 threads, 4 floats each
    int qi = qd[row], qai = qad[row];
    if (t < 32) {
        float4 v = ((const float4*)(qT + (size_t)qi * K_DIM))[t];
        v4h h = { (f16)v.x, (f16)v.y, (f16)v.z, (f16)v.w };
        *(v4h*)(qE + row * K_DIM + t * 4) = h;
    }
    {
        float4 v = ((const float4*)(qaT + (size_t)qai * QA_DIM))[t];
        v4h h = { (f16)v.x, (f16)v.y, (f16)v.z, (f16)v.w };
        *(v4h*)(qaE + row * QA_DIM + t * 4) = h;
    }
}

// ---------------- Kernel 2: w = softmax(qE @ key^T)  [SB x 64] ----------------
__global__ void kCorr(const f16* __restrict__ qE, const f16* __restrict__ keyh,
                      float* __restrict__ wOut) {
    int tid = threadIdx.x, lane = tid & 31, wv = tid >> 5;   // 128 thr = 4 waves
    int row0 = blockIdx.x * 16;
    int col0 = wv * 16;
    v8f c = {};
#pragma unroll
    for (int k0 = 0; k0 < K_DIM; k0 += 32) {
        v16h a = load_afrag(qE, K_DIM, row0, k0, lane);
        v16h b = load_bfragT(keyh, K_DIM, k0, col0, lane);
        c = wmma16(a, b, c);
    }
    __shared__ float lg[16 * M_SZ];
    int cl = lane & 15, rb = (lane >> 4) * 8;
#pragma unroll
    for (int i = 0; i < 8; ++i) lg[(rb + i) * M_SZ + col0 + cl] = c[i];
    __syncthreads();
    if (tid < 16) {
        float* rowp = lg + tid * M_SZ;
        float mx = -1e30f;
        for (int m = 0; m < M_SZ; ++m) mx = fmaxf(mx, rowp[m]);
        float sum = 0.f;
        for (int m = 0; m < M_SZ; ++m) { float e = __expf(rowp[m] - mx); rowp[m] = e; sum += e; }
        float inv = 1.0f / sum;
        float* o = wOut + (size_t)(row0 + tid) * M_SZ;
        for (int m = 0; m < M_SZ; ++m) o[m] = rowp[m] * inv;
    }
}

// ------- Kernel 3: e = sigmoid(qaE@eW + eb), a = tanh(qaE@aW + ab), f16 out -------
// 8 waves: even waves -> erase, odd -> add; each wave owns 64 cols (4 tiles), reuses A frag.
__global__ void kEraseAdd(const f16* __restrict__ qaE,
                          const f16* __restrict__ eWt, const float* __restrict__ eb,
                          const f16* __restrict__ aWt, const float* __restrict__ ab,
                          f16* __restrict__ eOut, f16* __restrict__ aOut) {
    int tid = threadIdx.x, lane = tid & 31, wv = tid >> 5;   // 256 thr = 8 waves
    int row0 = blockIdx.x * 16;
    bool isErase = (wv & 1) == 0;
    int colb = (wv >> 1) * 64;                               // 0,64,128,192
    const f16* Wt = isErase ? eWt : aWt;
    v8f c[4] = {};
#pragma unroll 4
    for (int k0 = 0; k0 < QA_DIM; k0 += 32) {
        v16h a = load_afrag(qaE, QA_DIM, row0, k0, lane);
#pragma unroll
        for (int t = 0; t < 4; ++t)
            c[t] = wmma16(a, load_bfragT(Wt, QA_DIM, k0, colb + t * 16, lane), c[t]);
    }
    int cl = lane & 15, rb = (lane >> 4) * 8;
    f16* Out = isErase ? eOut : aOut;
    const float* bias = isErase ? eb : ab;
#pragma unroll
    for (int t = 0; t < 4; ++t) {
        int cg = colb + t * 16 + cl;
        float bv = bias[cg];
#pragma unroll
        for (int i = 0; i < 8; ++i) {
            float x = c[t][i] + bv;
            float y = isErase ? (1.0f / (1.0f + __expf(-x))) : tanhf(x);
            Out[(size_t)(row0 + rb + i) * V_DIM + cg] = (f16)y;
        }
    }
}

// ---------------- Kernel 4: sequential scan, Mv in LDS ----------------
// grid (B, 4): block owns batch b, V-column slice [64*gy, 64*gy+64). 64 threads.
__global__ void kScan(const float* __restrict__ wBuf, const f16* __restrict__ eB,
                      const f16* __restrict__ aB, const float* __restrict__ Mv0,
                      f16* __restrict__ rB) {
    __shared__ float Mv[M_SZ * 64];   // 16 KB, lane->bank 1:1 on column access
    __shared__ float wS[M_SZ];
    int b = blockIdx.x;
    int vg = blockIdx.y * 64 + threadIdx.x;   // global V column
    int vl = threadIdx.x;                      // local V column
    for (int m = 0; m < M_SZ; ++m) Mv[m * 64 + vl] = Mv0[m * V_DIM + vg];
    __syncthreads();
    for (int s = 0; s < S_LEN; ++s) {
        size_t sb = (size_t)s * B_SZ + b;
        wS[vl] = wBuf[sb * M_SZ + vl];               // blockDim == 64 == M_SZ
        float ev = (float)eB[sb * V_DIM + vg];
        float av = (float)aB[sb * V_DIM + vg];
        __syncthreads();
        float r = 0.f;
#pragma unroll 8
        for (int m = 0; m < M_SZ; ++m) {
            float wm = wS[m];
            float mv = Mv[m * 64 + vl];
            r += wm * mv;                             // read uses pre-update Mv
            Mv[m * 64 + vl] = mv + wm * (av - ev * mv);
        }
        rB[((size_t)b * S_LEN + s) * V_DIM + vg] = (f16)r;   // [B,S,V] layout
        __syncthreads();
    }
}

// ------ Kernel 5: feat = tanh([reads|qE] @ read_W + rb); pred; loss partials ------
// 128 threads = 4 waves; each wave owns 32 cols (2 tiles), reuses A fragment.
__global__ void kFeat(const f16* __restrict__ rB, const f16* __restrict__ qE,
                      const f16* __restrict__ rWt, const float* __restrict__ rbias,
                      const float* __restrict__ pW, const float* __restrict__ pb,
                      const float* __restrict__ target,
                      float* __restrict__ outPred, float* __restrict__ partials) {
    int tid = threadIdx.x, lane = tid & 31, wv = tid >> 5;   // 128 thr = 4 waves
    int row0 = blockIdx.x * 16;
    int colb = wv * 32;
    int r = row0 + (lane & 15);                               // row = b*S + s
    int s = r & (S_LEN - 1);
    int b = r >> 10;
    const f16* rRow = rB + (size_t)r * V_DIM;                      // k in [0,256)
    const f16* qRow = qE + ((size_t)s * B_SZ + b) * K_DIM;         // k in [256,384)
    int kb = (lane >> 4) * 8;
    v8f c0 = {}, c1 = {};
#pragma unroll
    for (int k0 = 0; k0 < CAT_DIM; k0 += 32) {
        // chunk-level source select folds at compile time under full unroll
        const f16* p1 = (k0      < V_DIM) ? (rRow + k0)      : (qRow + (k0 - V_DIM));
        const f16* p2 = (k0 + 16 < V_DIM) ? (rRow + k0 + 16) : (qRow + (k0 + 16 - V_DIM));
        v16h a;
#pragma unroll
        for (int i = 0; i < 8; ++i) { a[i] = p1[kb + i]; a[8 + i] = p2[kb + i]; }
        c0 = wmma16(a, load_bfragT(rWt, CAT_DIM, k0, colb,      lane), c0);
        c1 = wmma16(a, load_bfragT(rWt, CAT_DIM, k0, colb + 16, lane), c1);
    }
    __shared__ float fS[16 * F_DIM];
    int cl = lane & 15, rb8 = (lane >> 4) * 8;
    {
        int cg0 = colb + cl;      float b0 = rbias[cg0];
        int cg1 = colb + 16 + cl; float b1 = rbias[cg1];
#pragma unroll
        for (int i = 0; i < 8; ++i) {
            fS[(rb8 + i) * F_DIM + cg0] = tanhf(c0[i] + b0);
            fS[(rb8 + i) * F_DIM + cg1] = tanhf(c1[i] + b1);
        }
    }
    __syncthreads();
    __shared__ float eAcc[16], mAcc[16];
    if (tid < 16) {
        float acc = pb[0];
        const float* fr = fS + tid * F_DIM;
        for (int j = 0; j < F_DIM; ++j) acc += fr[j] * pW[j];
        float x = 1.0f / (1.0f + __expf(-acc));   // pred (sigmoided)
        int rr = row0 + tid;
        outPred[rr] = x;
        float t = target[rr];
        float msk = (t >= 0.0f) ? 1.0f : 0.0f;
        // softplus(x) - x*t  (faithful: BCE-with-logits applied to sigmoided pred)
        float sp = (x > 0.f) ? (x + log1pf(__expf(-x))) : log1pf(__expf(x));
        eAcc[tid] = (sp - x * t) * msk;
        mAcc[tid] = msk;
    }
    __syncthreads();
    if (tid == 0) {
        float se = 0.f, sm = 0.f;
#pragma unroll
        for (int i = 0; i < 16; ++i) { se += eAcc[i]; sm += mAcc[i]; }
        partials[blockIdx.x] = se;
        partials[NROWT + blockIdx.x] = sm;
    }
}

// ---------------- Kernel 6: deterministic loss reduction ----------------
__global__ void kFinal(const float* __restrict__ partials, float* __restrict__ out) {
    if (threadIdx.x == 0 && blockIdx.x == 0) {
        float se = 0.f, sm = 0.f;
        for (int i = 0; i < NROWT; ++i) { se += partials[i]; sm += partials[NROWT + i]; }
        out[SB] = se / fmaxf(sm, 1.0f);
    }
}

extern "C" void kernel_launch(void* const* d_in, const int* in_sizes, int n_in,
                              void* d_out, int out_size, void* d_ws, size_t ws_size,
                              hipStream_t stream) {
    const int*   q_data  = (const int*)  d_in[0];
    const int*   qa_data = (const int*)  d_in[1];
    const float* target  = (const float*)d_in[2];
    const float* q_tab   = (const float*)d_in[3];
    const float* qa_tab  = (const float*)d_in[4];
    const float* key_mem = (const float*)d_in[5];
    const float* Mv0     = (const float*)d_in[6];
    const float* erase_W = (const float*)d_in[7];
    const float* erase_b = (const float*)d_in[8];
    const float* add_W   = (const float*)d_in[9];
    const float* add_b   = (const float*)d_in[10];
    const float* read_W  = (const float*)d_in[11];
    const float* read_b  = (const float*)d_in[12];
    const float* pred_W  = (const float*)d_in[13];
    const float* pred_b  = (const float*)d_in[14];
    float* out = (float*)d_out;

    char* ws = (char*)d_ws;
    f16*   qaE  = (f16*)ws;   ws += (size_t)SB * QA_DIM * sizeof(f16);
    f16*   qE   = (f16*)ws;   ws += (size_t)SB * K_DIM  * sizeof(f16);
    float* wBuf = (float*)ws; ws += (size_t)SB * M_SZ   * sizeof(float);
    f16*   eBuf = (f16*)ws;   ws += (size_t)SB * V_DIM  * sizeof(f16);
    f16*   aBuf = (f16*)ws;   ws += (size_t)SB * V_DIM  * sizeof(f16);
    f16*   rBuf = (f16*)ws;   ws += (size_t)SB * V_DIM  * sizeof(f16);
    float* part = (float*)ws; ws += (size_t)2 * NROWT   * sizeof(float);
    f16*   keyh = (f16*)ws;   ws += (size_t)M_SZ * K_DIM    * sizeof(f16);
    f16*   eWt  = (f16*)ws;   ws += (size_t)V_DIM * QA_DIM  * sizeof(f16);
    f16*   aWt  = (f16*)ws;   ws += (size_t)V_DIM * QA_DIM  * sizeof(f16);
    f16*   rWt  = (f16*)ws;   ws += (size_t)F_DIM * CAT_DIM * sizeof(f16);

    kPrepW<<<(V_DIM * QA_DIM + 255) / 256, 256, 0, stream>>>(
        key_mem, erase_W, add_W, read_W, keyh, eWt, aWt, rWt);
    kEmbed<<<SB, 64, 0, stream>>>(q_data, qa_data, q_tab, qa_tab, qE, qaE);
    kCorr<<<NROWT, 128, 0, stream>>>(qE, keyh, wBuf);
    kEraseAdd<<<NROWT, 256, 0, stream>>>(qaE, eWt, erase_b, aWt, add_b, eBuf, aBuf);
    kScan<<<dim3(B_SZ, 4), 64, 0, stream>>>(wBuf, eBuf, aBuf, Mv0, rBuf);
    kFeat<<<NROWT, 128, 0, stream>>>(rBuf, qE, rWt, read_b, pred_W, pred_b,
                                     target, out, part);
    kFinal<<<1, 32, 0, stream>>>(part, out);
}
